// BinaryDense_52347061404008
// MI455X (gfx1250) — matrix operand order repaired and checked by reading, not compiled, not run on hardware
//
#include <hip/hip_runtime.h>
#include <stdint.h>

#define TOKENS 8192
#define IN_CH  4096
#define OUT_CH 16384

#define BM 128
#define BN 128
#define BK 64
#define NK (IN_CH / BK)                 /* 64 K-tiles */
#define ROW_BYTES (BK * 2)              /* 128 B of bf16 per row per tile */
#define LDS_STRIDE 144                  /* 128 data + 16 pad: conflict-free b128 */
#define TILE_BYTES (128 * LDS_STRIDE)   /* 18432 B per operand tile */
#define BUF_BYTES (2 * TILE_BYTES)      /* A + B */
#define SMEM_BYTES (2 * BUF_BYTES)      /* double buffered: 73728 B */

static_assert(TOKENS % BM == 0 && OUT_CH % BN == 0 && IN_CH % BK == 0, "tiling");

typedef __attribute__((ext_vector_type(16))) __bf16 v16bf;
typedef __attribute__((ext_vector_type(8)))  float  v8f;

union FragAB { v16bf v; uint4 q[2]; };
static_assert(sizeof(FragAB) == 32, "frag size");

// ---------------- CDNA5 async global -> LDS copy (ASYNCcnt path) -------------
__device__ __forceinline__ void async_ld_b128(uint32_t lds_addr, const void* gptr) {
  // GV mode: 64-bit vaddr, saddr = off. VDST register carries the LDS byte address.
  asm volatile("global_load_async_to_lds_b128 %0, %1, off"
               :: "v"(lds_addr), "v"(gptr)
               : "memory");
}

// ---------------- f32 -> bf16 (round to nearest even) ------------------------
__device__ __forceinline__ uint16_t f2bf(float f) {
  union { float f; uint32_t u; } x; x.f = f;
  uint32_t u = x.u;
  u += 0x7FFFu + ((u >> 16) & 1u);
  return (uint16_t)(u >> 16);
}

// ---------------- prep: X f32 -> bf16 ----------------------------------------
__global__ void __launch_bounds__(256)
cvt_x_kernel(const float* __restrict__ x, uint16_t* __restrict__ xb, int n4) {
  int i = blockIdx.x * blockDim.x + threadIdx.x;
  if (i >= n4) return;
  float4 v = ((const float4*)x)[i];
  ushort4 r;
  r.x = f2bf(v.x); r.y = f2bf(v.y); r.z = f2bf(v.z); r.w = f2bf(v.w);
  ((ushort4*)xb)[i] = r;
}

// ---------------- prep: W -> binarized bf16 ----------------------------------
__global__ void __launch_bounds__(256)
cvt_w_kernel(const float* __restrict__ w, uint16_t* __restrict__ wb,
             const float* __restrict__ kk, int n4) {
  int i = blockIdx.x * blockDim.x + threadIdx.x;
  if (i >= n4) return;
  const float k = kk[0];
  const bool soft = (k < 1.0e6f);
  float4 v = ((const float4*)w)[i];
  float a0, a1, a2, a3;
  if (soft) {
    a0 = tanhf(v.x * k); a1 = tanhf(v.y * k);
    a2 = tanhf(v.z * k); a3 = tanhf(v.w * k);
  } else {
    a0 = (v.x > 0.f) ? 1.f : ((v.x < 0.f) ? -1.f : 0.f);
    a1 = (v.y > 0.f) ? 1.f : ((v.y < 0.f) ? -1.f : 0.f);
    a2 = (v.z > 0.f) ? 1.f : ((v.z < 0.f) ? -1.f : 0.f);
    a3 = (v.w > 0.f) ? 1.f : ((v.w < 0.f) ? -1.f : 0.f);
  }
  ushort4 r;
  r.x = f2bf(a0); r.y = f2bf(a1); r.z = f2bf(a2); r.w = f2bf(a3);
  ((ushort4*)wb)[i] = r;
}

// ---------------- main GEMM: D = X(bf16) * Wbin(bf16)^T + bias, scaled -------
__global__ void __launch_bounds__(256)
binary_dense_gemm(const uint16_t* __restrict__ xb,
                  const uint16_t* __restrict__ wb,
                  const float* __restrict__ bias,
                  const float* __restrict__ nmk,
                  float* __restrict__ out) {
  __shared__ char smem[SMEM_BYTES];
  // Flat address of an LDS object: high bits = shared aperture, low 32 bits =
  // raw LDS byte offset (ISA aperture rules). The ptrtoint also makes `smem`
  // escape so the compiler cannot prove "never stored" and delete the array.
  const uint32_t sbase = (uint32_t)(uintptr_t)(void*)smem;

  const int tid  = threadIdx.x;
  const int lane = tid & 31;
  const int wave = tid >> 5;
  const int wm   = wave & 3;      // 4 waves along M (tokens)
  const int wn   = wave >> 2;     // 2 waves along N (out channels)
  const int lr   = lane & 15;
  const int lh   = lane >> 4;     // lane half selects K sub-range
  const int t0   = blockIdx.y * BM;
  const int o0   = blockIdx.x * BN;

  // ---- staging assignment: each thread copies 64 consecutive bytes of one row
  const int srow = tid >> 1;            // 0..127
  const int scol = (tid & 1) * 64;      // 0 or 64 bytes within the 128B row
  const uint8_t* gA0 = (const uint8_t*)xb + ((size_t)(t0 + srow) * IN_CH) * 2 + scol;
  const uint8_t* gB0 = (const uint8_t*)wb + ((size_t)(o0 + srow) * IN_CH) * 2 + scol;
  const uint32_t lA0 = sbase + (uint32_t)(srow * LDS_STRIDE + scol);
  const uint32_t lB0 = sbase + (uint32_t)(TILE_BYTES + srow * LDS_STRIDE + scol);

  auto stage = [&](int kt, int buf) {
    const uint8_t* ga = gA0 + (size_t)kt * ROW_BYTES;
    const uint8_t* gb = gB0 + (size_t)kt * ROW_BYTES;
    const uint32_t base = (uint32_t)buf * BUF_BYTES;
#pragma unroll
    for (int c = 0; c < 4; ++c) {
      async_ld_b128(base + lA0 + c * 16, ga + c * 16);
      async_ld_b128(base + lB0 + c * 16, gb + c * 16);
    }
  };

  // ---- per-lane fragment base offsets (ISA 7.12.2 layouts)
  // A (16x32 bf16): lane = row M; lanes 0-15 take K{0-7,16-23} (+0,+32),
  //                 lanes 16-31 take K{8-15,24-31} (+16,+48)
  uint32_t aoff[2];
#pragma unroll
  for (int mi = 0; mi < 2; ++mi)
    aoff[mi] = sbase + (uint32_t)((wm * 32 + mi * 16 + lr) * LDS_STRIDE + lh * 16);
  // B (32x16 bf16): lane = col N (a W row); lanes 0-15 take K{0-7,16-23} (+0,+32),
  //                 lanes 16-31 take K{8-15,24-31} (+16,+48)
  uint32_t boff[4];
#pragma unroll
  for (int ni = 0; ni < 4; ++ni)
    boff[ni] = sbase + (uint32_t)(TILE_BYTES + (wn * 64 + ni * 16 + lr) * LDS_STRIDE + lh * 16);

  v8f acc[2][4] = {};

  stage(0, 0);
  for (int kt = 0; kt < NK; ++kt) {
    if (kt + 1 < NK) {
      stage(kt + 1, (kt + 1) & 1);
      asm volatile("s_wait_asynccnt 0x8" ::: "memory");  // current tile's 8 done
    } else {
      asm volatile("s_wait_asynccnt 0x0" ::: "memory");
    }
    __syncthreads();   // all waves' async LDS writes visible

    const uint32_t bb = (uint32_t)(kt & 1) * BUF_BYTES;
#pragma unroll
    for (int ks = 0; ks < 2; ++ks) {           // two K=32 steps per LDS tile
      const uint32_t ko = bb + (uint32_t)ks * 64;
      FragAB a[2], b[4];
      // Opaque LDS fragment loads: 12x ds_load_b128 + one dscnt wait in a
      // single asm block. The WMMA consumers depend on these outputs, so the
      // wait is ordered before any use and nothing can be dead-code-eliminated.
      asm volatile(
          "ds_load_b128 %0, %12\n\t"
          "ds_load_b128 %1, %12 offset:32\n\t"
          "ds_load_b128 %2, %13\n\t"
          "ds_load_b128 %3, %13 offset:32\n\t"
          "ds_load_b128 %4, %14\n\t"
          "ds_load_b128 %5, %14 offset:32\n\t"
          "ds_load_b128 %6, %15\n\t"
          "ds_load_b128 %7, %15 offset:32\n\t"
          "ds_load_b128 %8, %16\n\t"
          "ds_load_b128 %9, %16 offset:32\n\t"
          "ds_load_b128 %10, %17\n\t"
          "ds_load_b128 %11, %17 offset:32\n\t"
          "s_wait_dscnt 0x0"
          : "=v"(a[0].q[0]), "=v"(a[0].q[1]),
            "=v"(a[1].q[0]), "=v"(a[1].q[1]),
            "=v"(b[0].q[0]), "=v"(b[0].q[1]),
            "=v"(b[1].q[0]), "=v"(b[1].q[1]),
            "=v"(b[2].q[0]), "=v"(b[2].q[1]),
            "=v"(b[3].q[0]), "=v"(b[3].q[1])
          : "v"(ko + aoff[0]), "v"(ko + aoff[1]),
            "v"(ko + boff[0]), "v"(ko + boff[1]),
            "v"(ko + boff[2]), "v"(ko + boff[3])
          : "memory");
#pragma unroll
      for (int mi = 0; mi < 2; ++mi)
#pragma unroll
        for (int ni = 0; ni < 4; ++ni)
          acc[mi][ni] = __builtin_amdgcn_wmma_f32_16x16x32_bf16(
              false, a[mi].v, false, b[ni].v,
              (short)0, acc[mi][ni], false, false);
    }
    __syncthreads();   // done reading this buffer before it is re-filled
  }

  // ---- epilogue: nmk * (acc + bias). C/D layout: elem r -> M = r (+8 hi half),
  //      N = lane&15.
  const float s = nmk[0];
#pragma unroll
  for (int ni = 0; ni < 4; ++ni) {
    const int col = o0 + wn * 64 + ni * 16 + lr;
    const float bv = bias[col];
#pragma unroll
    for (int mi = 0; mi < 2; ++mi) {
      const int rbase = t0 + wm * 32 + mi * 16 + lh * 8;
#pragma unroll
      for (int r = 0; r < 8; ++r) {
        out[(size_t)(rbase + r) * OUT_CH + col] = s * (acc[mi][ni][r] + bv);
      }
    }
  }
}

// -----------------------------------------------------------------------------
extern "C" void kernel_launch(void* const* d_in, const int* in_sizes, int n_in,
                              void* d_out, int out_size, void* d_ws, size_t ws_size,
                              hipStream_t stream) {
  const float* x    = (const float*)d_in[0];
  const float* w    = (const float*)d_in[1];
  const float* bias = (const float*)d_in[2];
  const float* nmk  = (const float*)d_in[3];
  const float* kk   = (const float*)d_in[4];
  float* out = (float*)d_out;

  uint16_t* xb = (uint16_t*)d_ws;                       // 64 MB
  uint16_t* wb = xb + (size_t)TOKENS * IN_CH;           // 128 MB

  const int nx4 = TOKENS * IN_CH / 4;
  const int nw4 = OUT_CH * IN_CH / 4;
  cvt_x_kernel<<<nx4 / 256, 256, 0, stream>>>(x, xb, nx4);
  cvt_w_kernel<<<nw4 / 256, 256, 0, stream>>>(w, wb, kk, nw4);

  dim3 grid(OUT_CH / BN, TOKENS / BM);   // 128 x 64 blocks
  binary_dense_gemm<<<grid, 256, 0, stream>>>(xb, wb, bias, nmk, out);
}
